// ClusterMemory_47923245088802
// MI455X (gfx1250) — matrix operand order repaired and checked by reading, not compiled, not run on hardware
//
#include <hip/hip_runtime.h>
#include <hip/hip_bf16.h>
#include <math.h>

// ---------------------------------------------------------------------------
// Problem constants (from reference)
// ---------------------------------------------------------------------------
#define BATCH   4096
#define NSAMP   32768
#define DFEAT   1024
#define INV_T   20.0f          // 1 / 0.05
// Both operands are L2-normalized => |logit| <= 1/T = 20.  Fixed softmax max.
#define LOGIT_MAX 20.0f
#define M_BLOCK 32             // rows per workgroup (2 M-tiles)
#define N_SPLIT 2              // N range split across grid
#define N_HALF  (NSAMP / N_SPLIT)        // 16384 columns per block
#define N_QUADS (N_HALF / 64)            // 256 quads of 4 N-tiles per block

typedef __attribute__((ext_vector_type(16))) __bf16 v16bf;
typedef __attribute__((ext_vector_type(8)))  __bf16 v8bf;
typedef __attribute__((ext_vector_type(4)))  __bf16 v4bf;
typedef __attribute__((ext_vector_type(8)))  float  v8f;

union BF16x16 { v16bf v; v8bf h[2]; };

#define WMMA_BF16(A, B, C) __builtin_amdgcn_wmma_f32_16x16x32_bf16( \
    false, (A), false, (B), (short)0, (C), false, false)

// ---------------------------------------------------------------------------
// Kernel 0: zero the partial sum-exp accumulator S[BATCH]
// ---------------------------------------------------------------------------
__global__ __launch_bounds__(256) void k_zero(float* __restrict__ S) {
    S[blockIdx.x * 256 + threadIdx.x] = 0.0f;
}

// ---------------------------------------------------------------------------
// Kernel 1: row-wise L2 normalize inputs (f32) -> bf16
// ---------------------------------------------------------------------------
__global__ __launch_bounds__(256) void k_normalize(const float* __restrict__ in,
                                                   __bf16* __restrict__ out) {
    const int b   = blockIdx.x;
    const int tid = threadIdx.x;
    const float4 v = ((const float4*)(in + (size_t)b * DFEAT))[tid];
    float ss = v.x * v.x + v.y * v.y + v.z * v.z + v.w * v.w;
    #pragma unroll
    for (int d = 16; d > 0; d >>= 1) ss += __shfl_xor(ss, d, 32);
    __shared__ float sred[8];
    if ((tid & 31) == 0) sred[tid >> 5] = ss;
    __syncthreads();
    float tot = 0.0f;
    #pragma unroll
    for (int w = 0; w < 8; ++w) tot += sred[w];
    const float scale = 1.0f / fmaxf(sqrtf(tot), 1e-12f);
    v4bf r;
    r[0] = (__bf16)(v.x * scale);
    r[1] = (__bf16)(v.y * scale);
    r[2] = (__bf16)(v.z * scale);
    r[3] = (__bf16)(v.w * scale);
    *(v4bf*)(out + (size_t)b * DFEAT + tid * 4) = r;
}

// ---------------------------------------------------------------------------
// Kernel 2: f32 -> bf16 elementwise convert (features, 32K x 1024)
// ---------------------------------------------------------------------------
__global__ __launch_bounds__(256) void k_cvt_bf16(const float* __restrict__ in,
                                                  __bf16* __restrict__ out,
                                                  long long n4) {
    long long i      = blockIdx.x * (long long)blockDim.x + threadIdx.x;
    long long stride = gridDim.x * (long long)blockDim.x;
    for (; i < n4; i += stride) {
        const float4 v = ((const float4*)in)[i];
        v4bf r;
        r[0] = (__bf16)v.x; r[1] = (__bf16)v.y; r[2] = (__bf16)v.z; r[3] = (__bf16)v.w;
        ((v4bf*)out)[i] = r;
    }
}

// ---------------------------------------------------------------------------
// Kernel 3: fused GEMM (WMMA bf16) + fixed-max sum-exp partials.
// grid = 256 blocks = (128 row-blocks of 32 rows) x (2 N-halves).
// block = 256 threads = 8 waves.  Per wave iteration: 2 M-tiles x 4 N-tiles
// (8 WMMAs sharing 2 A fragments and 4 B tiles -> 1.5 loads per WMMA).
// Partials combine by atomicAdd (fixed softmax max => plain sums).
// ---------------------------------------------------------------------------
__global__ __launch_bounds__(256) void k_gemm_sumexp(const __bf16* __restrict__ xbf,
                                                     const __bf16* __restrict__ fbf,
                                                     float* __restrict__ S) {
    __shared__ __bf16 sX[M_BLOCK * DFEAT];     // 64 KB A-tile (32 rows)
    __shared__ float  sS[8 * M_BLOCK];

    const int tid    = threadIdx.x;
    const int rowblk = blockIdx.x >> 1;        // 0..127
    const int nhalf  = blockIdx.x & 1;         // which half of N

    // cooperative preload of the 32-row A block into LDS (4096 x b128)
    {
        const uint4* src = (const uint4*)(xbf + (size_t)rowblk * M_BLOCK * DFEAT);
        uint4*       dst = (uint4*)sX;
        #pragma unroll
        for (int i = 0; i < 16; ++i) dst[tid + 256 * i] = src[tid + 256 * i];
    }
    __syncthreads();

    const int wave = tid >> 5;
    const int lane = tid & 31;
    const int half = lane >> 4;                // 0: lanes 0-15, 1: lanes 16-31
    const int col  = lane & 15;

    // A-fragment bases: M-tile 0 = rows 0..15, M-tile 1 = rows 16..31
    const __bf16* sA0 = sX + (size_t)col * DFEAT + (half ? 16 : 0);
    const __bf16* sA1 = sA0 + 16 * DFEAT;

    // sum of exp(logit - 20) per slot v; rows: m-tile*16 + v + 8*half
    float sm0[8], sm1[8];
    #pragma unroll
    for (int v = 0; v < 8; ++v) { sm0[v] = 0.0f; sm1[v] = 0.0f; }

    const int nbase = nhalf * N_HALF;

    for (int q = wave; q < N_QUADS; q += 8) {
        const int n0 = nbase + q * 64 + col;   // this lane's column, tile 0
        const __bf16* bp = fbf + (size_t)n0 * DFEAT + (half ? 16 : 0);

        v8f a00 = {}, a01 = {}, a02 = {}, a03 = {};
        v8f a10 = {}, a11 = {}, a12 = {}, a13 = {};
        #pragma unroll 2
        for (int ks = 0; ks < 32; ++ks) {
            const int kb = ks * 32;
            BF16x16 fa0, fa1;
            fa0.h[0] = *(const v8bf*)(sA0 + kb);
            fa0.h[1] = *(const v8bf*)(sA0 + kb + 8);
            fa1.h[0] = *(const v8bf*)(sA1 + kb);
            fa1.h[1] = *(const v8bf*)(sA1 + kb + 8);
            const v16bf b0 = *(const v16bf*)(bp + kb);
            const v16bf b1 = *(const v16bf*)(bp + kb + 16 * DFEAT);
            const v16bf b2 = *(const v16bf*)(bp + kb + 32 * DFEAT);
            const v16bf b3 = *(const v16bf*)(bp + kb + 48 * DFEAT);
            a00 = WMMA_BF16(fa0.v, b0, a00);
            a10 = WMMA_BF16(fa1.v, b0, a10);
            a01 = WMMA_BF16(fa0.v, b1, a01);
            a11 = WMMA_BF16(fa1.v, b1, a11);
            a02 = WMMA_BF16(fa0.v, b2, a02);
            a12 = WMMA_BF16(fa1.v, b2, a12);
            a03 = WMMA_BF16(fa0.v, b3, a03);
            a13 = WMMA_BF16(fa1.v, b3, a13);
        }

        // logit = acc * 20 ; accumulate exp(logit - 20) = exp(20*acc - 20)
        #pragma unroll
        for (int v = 0; v < 8; ++v) {
            sm0[v] += __expf(fmaf(a00[v], INV_T, -LOGIT_MAX));
            sm0[v] += __expf(fmaf(a01[v], INV_T, -LOGIT_MAX));
            sm0[v] += __expf(fmaf(a02[v], INV_T, -LOGIT_MAX));
            sm0[v] += __expf(fmaf(a03[v], INV_T, -LOGIT_MAX));
            sm1[v] += __expf(fmaf(a10[v], INV_T, -LOGIT_MAX));
            sm1[v] += __expf(fmaf(a11[v], INV_T, -LOGIT_MAX));
            sm1[v] += __expf(fmaf(a12[v], INV_T, -LOGIT_MAX));
            sm1[v] += __expf(fmaf(a13[v], INV_T, -LOGIT_MAX));
        }
    }

    // sum-reduce across the 16 lanes of each half
    #pragma unroll
    for (int d = 1; d < 16; d <<= 1) {
        #pragma unroll
        for (int v = 0; v < 8; ++v) {
            sm0[v] += __shfl_xor(sm0[v], d, 32);
            sm1[v] += __shfl_xor(sm1[v], d, 32);
        }
    }

    if (col == 0) {
        #pragma unroll
        for (int v = 0; v < 8; ++v) {
            sS[wave * M_BLOCK + v + 8 * half]      = sm0[v];
            sS[wave * M_BLOCK + 16 + v + 8 * half] = sm1[v];
        }
    }
    __syncthreads();

    // combine across the 8 waves; add this block's partial into global S
    if (tid < M_BLOCK) {
        float s = 0.0f;
        #pragma unroll
        for (int w = 0; w < 8; ++w) s += sS[w * M_BLOCK + tid];
        atomicAdd(&S[rowblk * M_BLOCK + tid], s);
    }
}

// ---------------------------------------------------------------------------
// Kernel 4: target logit per row: dot(x̂_b, f_t)/T, one wave per row
// ---------------------------------------------------------------------------
__global__ __launch_bounds__(256) void k_target_dot(const __bf16* __restrict__ xbf,
                                                    const __bf16* __restrict__ fbf,
                                                    const int* __restrict__ tgt,
                                                    float* __restrict__ tl) {
    const int b    = blockIdx.x * 8 + (threadIdx.x >> 5);
    const int lane = threadIdx.x & 31;
    const int t    = tgt[b];
    const __bf16* xr = xbf + (size_t)b * DFEAT;
    const __bf16* fr = fbf + (size_t)t * DFEAT;
    float s = 0.0f;
    for (int k = lane * 4; k < DFEAT; k += 128) {
        const v4bf xv = *(const v4bf*)(xr + k);
        const v4bf fv = *(const v4bf*)(fr + k);
        s += (float)xv[0] * (float)fv[0] + (float)xv[1] * (float)fv[1] +
             (float)xv[2] * (float)fv[2] + (float)xv[3] * (float)fv[3];
    }
    #pragma unroll
    for (int d = 16; d > 0; d >>= 1) s += __shfl_xor(s, d, 32);
    if (lane == 0) tl[b] = s * INV_T;
}

// ---------------------------------------------------------------------------
// Kernel 5: mean over rows of (20 + log(S[b]) - target_logit[b])
// ---------------------------------------------------------------------------
__global__ __launch_bounds__(256) void k_mean(const float* __restrict__ S,
                                              const float* __restrict__ tl,
                                              float* __restrict__ out) {
    __shared__ float red[256];
    float s = 0.0f;
    for (int b = threadIdx.x; b < BATCH; b += 256)
        s += (LOGIT_MAX + __logf(S[b])) - tl[b];
    red[threadIdx.x] = s;
    __syncthreads();
    #pragma unroll
    for (int k = 128; k > 0; k >>= 1) {
        if (threadIdx.x < k) red[threadIdx.x] += red[threadIdx.x + k];
        __syncthreads();
    }
    if (threadIdx.x == 0) out[0] = red[0] * (1.0f / (float)BATCH);
}

// ---------------------------------------------------------------------------
// Launch
// ---------------------------------------------------------------------------
extern "C" void kernel_launch(void* const* d_in, const int* in_sizes, int n_in,
                              void* d_out, int out_size, void* d_ws, size_t ws_size,
                              hipStream_t stream) {
    (void)in_sizes; (void)n_in; (void)out_size; (void)ws_size;

    const float* inputs = (const float*)d_in[0];   // [4096,1024] f32
    const int*   ctgt   = (const int*)d_in[2];     // corrected_targets [4096]
    const float* feats  = (const float*)d_in[3];   // [32768,1024] f32

    char* ws = (char*)d_ws;
    __bf16* xbf = (__bf16*)(ws);                                   //  8 MB
    __bf16* fbf = (__bf16*)(ws + ((size_t)8  << 20));              // 64 MB
    float*  S   = (float*)(ws + ((size_t)72 << 20));               // 16 KB
    float*  tl  = (float*)(ws + ((size_t)72 << 20) + (16 << 10));  // 16 KB

    k_zero      <<<BATCH / 256, 256, 0, stream>>>(S);
    k_normalize <<<BATCH, 256, 0, stream>>>(inputs, xbf);
    k_cvt_bf16  <<<8192, 256, 0, stream>>>(feats, fbf,
                                           (long long)NSAMP * DFEAT / 4);
    k_gemm_sumexp<<<(BATCH / M_BLOCK) * N_SPLIT, 256, 0, stream>>>(xbf, fbf, S);
    k_target_dot<<<BATCH / 8, 256, 0, stream>>>(xbf, fbf, ctgt, tl);
    k_mean      <<<1, 256, 0, stream>>>(S, tl, (float*)d_out);
}
